// SelfAttention_91285234909786
// MI455X (gfx1250) — matrix operand order repaired and checked by reading, compiled, not run on hardware
//
#include <hip/hip_runtime.h>
#include <hip/hip_bf16.h>

typedef __bf16 bf16_t;
typedef __attribute__((ext_vector_type(16))) __bf16 v16bf;
typedef __attribute__((ext_vector_type(8)))  __bf16 v8bf;
typedef __attribute__((ext_vector_type(8)))  float  v8f;
typedef __attribute__((ext_vector_type(4)))  float  v4f;   // native vector for NT stores

// Problem constants (reference: B=2, S=2048, D=1024, H=16, HD=64)
#define BB 2
#define SS 2048
#define DD 1024
#define HH 16
#define HD 64
#define QK_SCALE 0.125f   // 64^-0.5

#define SC_STRIDE 2056    // 2048 + 8 pad floats -> spread LDS banks on tile reads

// ---------- helpers ----------
static __device__ __forceinline__ v16bf ldfrag(const bf16_t* p0, const bf16_t* p1) {
  v8bf lo = *(const v8bf*)p0;
  v8bf hi = *(const v8bf*)p1;
  return __builtin_shufflevector(lo, hi, 0,1,2,3,4,5,6,7,8,9,10,11,12,13,14,15);
}

static __device__ __forceinline__ v8f wmma_bf16(v16bf a, v16bf b, v8f c) {
  return __builtin_amdgcn_wmma_f32_16x16x32_bf16(false, a, false, b, (short)0, c, false, false);
}

// ---------- kernel 1: f32 -> bf16 conversion ----------
__global__ void cvt_f32_bf16(const float* __restrict__ src, bf16_t* __restrict__ dst, int n) {
  int i = (blockIdx.x * blockDim.x + threadIdx.x) * 4;
  if (i + 3 < n) {
    float4 v = *(const float4*)(src + i);
    dst[i + 0] = (bf16_t)v.x;
    dst[i + 1] = (bf16_t)v.y;
    dst[i + 2] = (bf16_t)v.z;
    dst[i + 3] = (bf16_t)v.w;
  }
}

// ---------- kernel 2: fused QKV projection (bf16 WMMA, f32 accum) ----------
// Each wave computes a 64x64 tile of one of {Q, K, V} = x @ W^T.
// Q,K stored [B,H,S,HD]; V stored transposed [B,H,HD,S].
__global__ __launch_bounds__(256) void qkv_gemm(
    const bf16_t* __restrict__ xb,
    const bf16_t* __restrict__ wq, const bf16_t* __restrict__ wk, const bf16_t* __restrict__ wv,
    bf16_t* __restrict__ Q, bf16_t* __restrict__ Kd, bf16_t* __restrict__ VT)
{
  const int wave = blockIdx.x * 8 + (threadIdx.x >> 5);
  const int lane = threadIdx.x & 31;
  const int mat  = wave / (64 * 16);        // 0=Q, 1=K, 2=V
  const int rem  = wave % (64 * 16);
  const int m0   = (rem / 16) * 64;         // token-tile base (over B*S = 4096)
  const int n0   = (rem % 16) * 64;         // output-channel base (over D = 1024)
  const bf16_t* W = (mat == 0) ? wq : (mat == 1) ? wk : wv;

  const int r  = lane & 15;
  const int ah = lane >> 4;

  v8f acc[4][4] = {};
  for (int kb = 0; kb < DD; kb += 32) {
    v16bf a[4];
#pragma unroll
    for (int ma = 0; ma < 4; ++ma) {
      const bf16_t* ap = xb + (size_t)(m0 + ma * 16 + r) * DD + kb + ah * 8;
      a[ma] = ldfrag(ap, ap + 16);
    }
#pragma unroll
    for (int nb = 0; nb < 4; ++nb) {
      const bf16_t* bp = W + (size_t)(n0 + nb * 16 + r) * DD + kb + ah * 8;
      v16bf bfr = ldfrag(bp, bp + 16);
#pragma unroll
      for (int ma = 0; ma < 4; ++ma)
        acc[ma][nb] = wmma_bf16(a[ma], bfr, acc[ma][nb]);
    }
  }

  const float scale = (mat == 0) ? QK_SCALE : 1.0f;
#pragma unroll
  for (int nb = 0; nb < 4; ++nb) {
    const int n  = n0 + nb * 16 + r;
    const int h  = n >> 6;
    const int hd = n & 63;
#pragma unroll
    for (int ma = 0; ma < 4; ++ma) {
#pragma unroll
      for (int i = 0; i < 8; ++i) {
        const int m   = m0 + ma * 16 + ah * 8 + i;
        const int bi  = m >> 11;            // batch
        const int sIn = m & (SS - 1);       // token in batch
        const bf16_t val = (bf16_t)(acc[ma][nb][i] * scale);
        if (mat == 2) {
          VT[(((size_t)bi * HH + h) * HD + hd) * SS + sIn] = val;
        } else {
          bf16_t* dst = (mat == 0) ? Q : Kd;
          dst[(((size_t)bi * HH + h) * SS + sIn) * HD + hd] = val;
        }
      }
    }
  }
}

// ---------- kernel 3: attention core ----------
// One 4-wave workgroup per (b, h, 16-query block). Scores live in LDS for a
// single-pass softmax; normalized weights streamed to d_out with NT stores;
// P*V done with bf16 WMMA on LDS-resident P.
__global__ __launch_bounds__(128) void attn_kernel(
    const bf16_t* __restrict__ Q, const bf16_t* __restrict__ Kd, const bf16_t* __restrict__ VT,
    const float* __restrict__ bias, float* __restrict__ attnW, bf16_t* __restrict__ ctx)
{
  extern __shared__ float smem[];
  float* sc    = smem;                              // [16][SC_STRIDE]
  float* red   = smem + 16 * SC_STRIDE;             // [4][16][64] PV partials
  float* pstat = red + 4 * 16 * 64;                 // [16][8] partial max/sum
  float* rstat = pstat + 16 * 8;                    // [16] row max / inv-sum

  const int q0   = blockIdx.x * 16;
  const int h    = blockIdx.y;
  const int b    = blockIdx.z;
  const int tid  = threadIdx.x;
  const int w    = tid >> 5;
  const int lane = tid & 31;
  const int r    = lane & 15;
  const int ah   = lane >> 4;

  const bf16_t* Qh = Q  + ((size_t)b * HH + h) * SS * HD;
  const bf16_t* Kh = Kd + ((size_t)b * HH + h) * SS * HD;
  const bf16_t* Vh = VT + ((size_t)b * HH + h) * HD * SS;

  // --- Phase 1: raw scores, waves split the 128 key tiles ---
  v16bf aq0, aq1;
  {
    const bf16_t* qp = Qh + (size_t)(q0 + r) * HD;
    aq0 = ldfrag(qp + ah * 8,      qp + ah * 8 + 16);
    aq1 = ldfrag(qp + 32 + ah * 8, qp + 32 + ah * 8 + 16);
  }
  for (int kt = w * 32; kt < w * 32 + 32; ++kt) {
    v8f acc = {};
    const bf16_t* kp = Kh + (size_t)(kt * 16 + r) * HD + ah * 8;
    acc = wmma_bf16(aq0, ldfrag(kp,      kp + 16), acc);
    acc = wmma_bf16(aq1, ldfrag(kp + 32, kp + 48), acc);
#pragma unroll
    for (int i = 0; i < 8; ++i)
      sc[(ah * 8 + i) * SC_STRIDE + kt * 16 + r] = acc[i];
  }
  __syncthreads();

  // --- Phase 2: add bias, partial row max (8 threads per row, 256 cols each) ---
  {
    const int row = tid >> 3, ch = tid & 7;
    const float* bp = bias + ((size_t)b * SS + q0 + row) * SS + ch * 256;
    float* sp = sc + row * SC_STRIDE + ch * 256;
    float m = -3.402823466e38f;
    for (int j = 0; j < 256; j += 4) {
      float4 bv = *(const float4*)(bp + j);
      float4 sv = *(float4*)(sp + j);
      sv.x += bv.x; sv.y += bv.y; sv.z += bv.z; sv.w += bv.w;
      *(float4*)(sp + j) = sv;
      m = fmaxf(m, fmaxf(fmaxf(sv.x, sv.y), fmaxf(sv.z, sv.w)));
    }
    pstat[row * 8 + ch] = m;
  }
  __syncthreads();
  if (tid < 16) {
    float m = pstat[tid * 8];
    for (int j = 1; j < 8; ++j) m = fmaxf(m, pstat[tid * 8 + j]);
    rstat[tid] = m;
  }
  __syncthreads();

  // --- Phase 3: exp + partial row sum ---
  {
    const int row = tid >> 3, ch = tid & 7;
    const float m = rstat[row];
    float* sp = sc + row * SC_STRIDE + ch * 256;
    float sum = 0.f;
    for (int j = 0; j < 256; ++j) {
      float p = __expf(sp[j] - m);
      sp[j] = p;
      sum += p;
    }
    pstat[row * 8 + ch] = sum;
  }
  __syncthreads();
  if (tid < 16) {
    float s = 0.f;
    for (int j = 0; j < 8; ++j) s += pstat[tid * 8 + j];
    rstat[tid] = 1.0f / s;
  }
  __syncthreads();

  // --- Phase 4: normalize in LDS + stream attn_weights to HBM (non-temporal) ---
  {
    const int row = tid >> 3, ch = tid & 7;
    const float inv = rstat[row];
    float* sp = sc + row * SC_STRIDE + ch * 256;
    float* op = attnW + (((size_t)b * HH + h) * SS + q0 + row) * SS + ch * 256;
    for (int j = 0; j < 256; j += 4) {
      v4f sv = *(v4f*)(sp + j);
      sv *= inv;
      *(v4f*)(sp + j) = sv;
      __builtin_nontemporal_store(sv, (v4f*)(op + j));
    }
  }
  __syncthreads();

  // --- Phase 5: P @ V, waves split K = 2048 into 4x512 ---
  v8f oacc[4] = {};
  for (int kb = w * 512; kb < w * 512 + 512; kb += 32) {
    const float* sp = sc + r * SC_STRIDE + kb;
    float tmp[16];
    *(v4f*)(tmp + 0)  = *(const v4f*)(sp + ah * 8);
    *(v4f*)(tmp + 4)  = *(const v4f*)(sp + ah * 8 + 4);
    *(v4f*)(tmp + 8)  = *(const v4f*)(sp + 16 + ah * 8);
    *(v4f*)(tmp + 12) = *(const v4f*)(sp + 16 + ah * 8 + 4);
    v16bf a;
#pragma unroll
    for (int i = 0; i < 16; ++i) a[i] = (bf16_t)tmp[i];
#pragma unroll
    for (int nb = 0; nb < 4; ++nb) {
      const bf16_t* vp = Vh + (size_t)(nb * 16 + r) * SS + kb + ah * 8;
      oacc[nb] = wmma_bf16(a, ldfrag(vp, vp + 16), oacc[nb]);
    }
  }
#pragma unroll
  for (int nb = 0; nb < 4; ++nb)
#pragma unroll
    for (int i = 0; i < 8; ++i)
      red[(w * 16 + ah * 8 + i) * 64 + nb * 16 + r] = oacc[nb][i];
  __syncthreads();

  // --- Phase 6: cross-wave reduce, store bf16 context [B,S,D] ---
  for (int e = tid; e < 16 * 64; e += 128) {
    const int row = e >> 6, col = e & 63;
    float s = red[row * 64 + col] + red[(16 + row) * 64 + col] +
              red[(32 + row) * 64 + col] + red[(48 + row) * 64 + col];
    ctx[((size_t)b * SS + q0 + row) * DD + h * HD + col] = (bf16_t)s;
  }
}

// ---------- kernel 4: output projection (ctx @ Wo^T), f32 out ----------
__global__ __launch_bounds__(256) void out_proj(
    const bf16_t* __restrict__ ctx, const bf16_t* __restrict__ wo, float* __restrict__ out)
{
  const int wave = blockIdx.x * 8 + (threadIdx.x >> 5);
  const int lane = threadIdx.x & 31;
  const int m0   = (wave / 16) * 64;
  const int n0   = (wave % 16) * 64;
  const int r    = lane & 15;
  const int ah   = lane >> 4;

  v8f acc[4][4] = {};
  for (int kb = 0; kb < DD; kb += 32) {
    v16bf a[4];
#pragma unroll
    for (int ma = 0; ma < 4; ++ma) {
      const bf16_t* ap = ctx + (size_t)(m0 + ma * 16 + r) * DD + kb + ah * 8;
      a[ma] = ldfrag(ap, ap + 16);
    }
#pragma unroll
    for (int nb = 0; nb < 4; ++nb) {
      const bf16_t* bp = wo + (size_t)(n0 + nb * 16 + r) * DD + kb + ah * 8;
      v16bf bfr = ldfrag(bp, bp + 16);
#pragma unroll
      for (int ma = 0; ma < 4; ++ma)
        acc[ma][nb] = wmma_bf16(a[ma], bfr, acc[ma][nb]);
    }
  }
#pragma unroll
  for (int nb = 0; nb < 4; ++nb)
#pragma unroll
    for (int ma = 0; ma < 4; ++ma)
#pragma unroll
      for (int i = 0; i < 8; ++i)
        out[(size_t)(m0 + ma * 16 + ah * 8 + i) * DD + n0 + nb * 16 + r] = acc[ma][nb][i];
}

// ---------- host launch ----------
extern "C" void kernel_launch(void* const* d_in, const int* in_sizes, int n_in,
                              void* d_out, int out_size, void* d_ws, size_t ws_size,
                              hipStream_t stream) {
  (void)in_sizes; (void)n_in; (void)out_size; (void)ws_size;
  const float* x    = (const float*)d_in[0];
  const float* bias = (const float*)d_in[1];
  const float* Wq   = (const float*)d_in[2];
  const float* Wk   = (const float*)d_in[3];
  const float* Wv   = (const float*)d_in[4];
  const float* Wo   = (const float*)d_in[5];

  const size_t nX = (size_t)BB * SS * DD;   // 4,194,304
  const size_t nW = (size_t)DD * DD;        // 1,048,576

  // Workspace layout (bf16): x | Wq | Wk | Wv | Wo | Q | K | V^T | ctx  (~48 MB)
  bf16_t* xb  = (bf16_t*)d_ws;
  bf16_t* wqb = xb  + nX;
  bf16_t* wkb = wqb + nW;
  bf16_t* wvb = wkb + nW;
  bf16_t* wob = wvb + nW;
  bf16_t* Qb  = wob + nW;
  bf16_t* Kb  = Qb  + nX;
  bf16_t* VTb = Kb  + nX;
  bf16_t* ctx = VTb + nX;

  float* attn_out = (float*)d_out;          // [B,S,D]
  float* attnW    = attn_out + nX;          // [B,H,S,S]

  // 1) convert to bf16
  cvt_f32_bf16<<<(int)((nX / 4 + 255) / 256), 256, 0, stream>>>(x,  xb,  (int)nX);
  cvt_f32_bf16<<<(int)((nW / 4 + 255) / 256), 256, 0, stream>>>(Wq, wqb, (int)nW);
  cvt_f32_bf16<<<(int)((nW / 4 + 255) / 256), 256, 0, stream>>>(Wk, wkb, (int)nW);
  cvt_f32_bf16<<<(int)((nW / 4 + 255) / 256), 256, 0, stream>>>(Wv, wvb, (int)nW);
  cvt_f32_bf16<<<(int)((nW / 4 + 255) / 256), 256, 0, stream>>>(Wo, wob, (int)nW);

  // 2) fused QKV projection: 3 * 64 * 16 = 3072 waves, 8 waves/block
  qkv_gemm<<<384, 256, 0, stream>>>(xb, wqb, wkb, wvb, Qb, Kb, VTb);

  // 3) attention: one WG per (16 queries, head, batch); ~148 KB dynamic LDS
  const size_t smem = (size_t)(16 * SC_STRIDE + 4 * 16 * 64 + 16 * 8 + 16) * sizeof(float);
  attn_kernel<<<dim3(SS / 16, HH, BB), 128, smem, stream>>>(Qb, Kb, VTb, bias, attnW, ctx);

  // 4) output projection: 64 * 16 = 1024 waves, 8 waves/block
  out_proj<<<128, 256, 0, stream>>>(ctx, wob, attn_out);
}
